// CharRnn_38938173506137
// MI455X (gfx1250) — matrix operand order repaired and checked by reading, not compiled, not run on hardware
//
#include <hip/hip_runtime.h>

// ---------------- problem dims ----------------
#define NL 256      // n_letters
#define ED 1024     // embed dim
#define HD 2048     // hidden dim
#define SL 4096     // seq len
#define GD (4*HD)   // 8192 gate rows

// ---------------- feature gates (graceful fallback if builtins absent) ----------------
#ifndef USE_ASYNC_COPY
#define USE_ASYNC_COPY 1
#endif
#ifndef USE_TDM
#define USE_TDM 1
#endif

#if USE_ASYNC_COPY && __has_builtin(__builtin_amdgcn_global_load_async_to_lds_b128)
#define HAVE_ASYNC 1
#else
#define HAVE_ASYNC 0
#endif

#if HAVE_ASYNC
#if __has_builtin(__builtin_amdgcn_s_wait_asynccnt)
#define WAIT_ASYNC() __builtin_amdgcn_s_wait_asynccnt(0)
#else
#define WAIT_ASYNC() asm volatile("s_wait_asynccnt 0" ::: "memory")
#endif
#else
#define WAIT_ASYNC()
#endif

#if USE_TDM && __has_builtin(__builtin_amdgcn_tensor_load_to_lds) && \
    __has_builtin(__builtin_amdgcn_s_wait_tensorcnt)
#define HAVE_TDM 1
#else
#define HAVE_TDM 0
#endif

// ---------------- types ----------------
typedef __bf16 bf16x4 __attribute__((ext_vector_type(4)));
typedef __bf16 bf16x8 __attribute__((ext_vector_type(8)));
typedef __bf16 v16bf  __attribute__((ext_vector_type(16)));
typedef float  v8f    __attribute__((ext_vector_type(8)));
typedef unsigned uint4v __attribute__((ext_vector_type(4)));
typedef int      int8v  __attribute__((ext_vector_type(8)));
typedef int      int4v  __attribute__((ext_vector_type(4)));

// async-copy builtin wants V4i pointers in global (AS1) / LDS (AS3) address spaces
typedef __attribute__((address_space(1))) int4v* gv4p_t;
typedef __attribute__((address_space(3))) int4v* lv4p_t;
__device__ __forceinline__ gv4p_t as_g4(const void* p) { return (gv4p_t)(uintptr_t)p; }
__device__ __forceinline__ lv4p_t as_l4(void* p) { return (lv4p_t)(unsigned)(uintptr_t)p; }

__device__ __forceinline__ __bf16 to_bf16(float f) {
  unsigned u = __builtin_bit_cast(unsigned, f);
  u = (u + 0x7FFFu + ((u >> 16) & 1u)) >> 16;            // round-to-nearest-even
  unsigned short s = (unsigned short)u;
  return __builtin_bit_cast(__bf16, s);
}
__device__ __forceinline__ float sigmoid_f(float x) { return 1.0f / (1.0f + __expf(-x)); }

// ---------------- prep kernels ----------------
__global__ void k_bias_sum(const float* __restrict__ a, const float* __restrict__ b,
                           float* __restrict__ o) {
  int i = blockIdx.x * blockDim.x + threadIdx.x;
  if (i < GD) o[i] = a[i] + b[i];
}

__global__ void k_cvt_bf16(const float* __restrict__ src, __bf16* __restrict__ dst, int n) {
  int i = blockIdx.x * blockDim.x + threadIdx.x;
  int stride = gridDim.x * blockDim.x;
  for (; i < n; i += stride) dst[i] = to_bf16(src[i]);
}

__global__ void k_embed(const int* __restrict__ ids, const float* __restrict__ Wemb,
                        __bf16* __restrict__ emb) {
  int t = blockIdx.x;
  int id = ids[t];
  const float* src = Wemb + (size_t)id * ED;
  __bf16* dst = emb + (size_t)t * ED;
  for (int i = threadIdx.x; i < ED; i += blockDim.x) dst[i] = to_bf16(src[i]);
}

// ---------------- bf16 WMMA GEMM:  C[M,N] = A[M,K] * B[N,K]^T + bias[N] ----------------
// Block = 256 threads = 8 waves (4M x 2N), block tile 64x128, k-chunk 32.
// A/B k-tiles double-buffered in LDS, filled by async global->LDS DMA
// (ASYNCcnt + s_wait_asynccnt) overlapping the WMMA clause of the live buffer.
// All fragments are pulled from LDS into registers first so the 4 WMMAs issue
// back-to-back behind a single s_wait_dscnt.
#define BM 64
#define BN 128
#define BK 32

__global__ void k_gemm_bf16(float* __restrict__ C, const __bf16* __restrict__ A,
                            const __bf16* __restrict__ B, const float* __restrict__ bias,
                            int K, int lda, int ldb, int ldc) {
  __shared__ __bf16 Ash[2][BM * BK];    // 2 x 4KB
  __shared__ __bf16 Bsh[2][BN * BK];    // 2 x 8KB
  const int tid  = threadIdx.x;
  const int lane = tid & 31;
  const int w    = tid >> 5;
  const int wm   = w & 3;               // 0..3 (M)
  const int wn   = w >> 2;              // 0..1 (N)
  const int m0   = blockIdx.y * BM;
  const int n0   = blockIdx.x * BN;
  const int lrow = lane & 15;
  const int hs   = lane >> 4;

  const int arow = tid >> 2, aseg = tid & 3;   // A tile: 64 rows x 4 x 16B

  v8f acc[4];
#pragma unroll
  for (int t = 0; t < 4; ++t)
#pragma unroll
    for (int r = 0; r < 8; ++r) acc[t][r] = 0.0f;

  auto stage = [&](int buf, int k0) {
    {
      const __bf16* ga = A + (size_t)(m0 + arow) * lda + k0 + aseg * 8;
      __bf16* la = &Ash[buf][arow * BK + aseg * 8];
#if HAVE_ASYNC
      __builtin_amdgcn_global_load_async_to_lds_b128(as_g4(ga), as_l4(la), 0, 0);
#else
      *(bf16x8*)la = *(const bf16x8*)ga;
#endif
    }
#pragma unroll
    for (int rr = 0; rr < 2; ++rr) {
      const int idx  = tid + rr * 256;
      const int brow = idx >> 2, bseg = idx & 3;
      const __bf16* gb = B + (size_t)(n0 + brow) * ldb + k0 + bseg * 8;
      __bf16* lb = &Bsh[buf][brow * BK + bseg * 8];
#if HAVE_ASYNC
      __builtin_amdgcn_global_load_async_to_lds_b128(as_g4(gb), as_l4(lb), 0, 0);
#else
      *(bf16x8*)lb = *(const bf16x8*)gb;
#endif
    }
  };

  const int nk = K / BK;
  stage(0, 0);
  WAIT_ASYNC();
  __syncthreads();

  union frag { v16bf v; bf16x8 h[2]; };

  for (int kt = 0; kt < nk; ++kt) {
    const int cur = kt & 1;
    if (kt + 1 < nk) stage(cur ^ 1, (kt + 1) * BK);

    // Pull A fragment + all 4 B fragments into registers first...
    const __bf16* Ab = &Ash[cur][(wm * 16 + lrow) * BK + hs * 8];
    frag af;
    af.h[0] = *(const bf16x8*)(Ab);
    af.h[1] = *(const bf16x8*)(Ab + 16);
    frag bfr[4];
#pragma unroll
    for (int t = 0; t < 4; ++t) {
      const __bf16* Bb = &Bsh[cur][(wn * 64 + t * 16 + lrow) * BK + hs * 8];
      bfr[t].h[0] = *(const bf16x8*)(Bb);
      bfr[t].h[1] = *(const bf16x8*)(Bb + 16);
    }
    // ...then issue the WMMA clause back-to-back.
#pragma unroll
    for (int t = 0; t < 4; ++t)
      acc[t] = __builtin_amdgcn_wmma_f32_16x16x32_bf16(
          false, af.v, false, bfr[t].v, (short)0, acc[t], false, false);

    WAIT_ASYNC();
    __syncthreads();
  }

#pragma unroll
  for (int t = 0; t < 4; ++t) {
    const int n = n0 + wn * 64 + t * 16 + lrow;
    const float bv = bias ? bias[n] : 0.0f;
#pragma unroll
    for (int r = 0; r < 8; ++r) {
      const int m = m0 + wm * 16 + hs * 8 + r;
      C[(size_t)m * ldc + n] = acc[t][r] + bv;
    }
  }
}

// ---------------- persistent LSTM recurrence ----------------
// Grid = 128 WGs x 512 threads (16 waves). Wave (b,w) owns h-index j = b*16+w.
// Its 4 W_hh gate rows (i,f,g,o; bf16) stay resident in LDS for all 4096 steps:
// 16 waves * 4 rows * 2048 * 2B = 256KB + 8KB h staging <= 320KB WGP LDS.
// The one-time fill is a TDM job: one tensor_load_to_lds per wave, tile 2048x4 bf16,
// line stride HD*HD elements, tracked by TENSORcnt.
#define LSTM_WGS   128
#define LSTM_THR   512
#define LSTM_LDS   (16 * 4 * HD * 2 /*bf16 W*/ + HD * 4 /*f32 h*/)   // 270336 B

__global__ void k_lstm(const float* __restrict__ Whh, const __bf16* __restrict__ Whh_bf,
                       const float* __restrict__ xgates,
                       const float* __restrict__ h0, const float* __restrict__ c0,
                       float* __restrict__ hbuf /*[2][HD]*/, __bf16* __restrict__ hs_bf,
                       unsigned* __restrict__ bar, float* __restrict__ out_tail) {
  extern __shared__ char lds_raw[];
  __bf16* wlds = (__bf16*)lds_raw;                 // [16 waves][4 rows][HD]
  float*  hsh  = (float*)(lds_raw + 16 * 4 * HD * 2);

  const int tid  = threadIdx.x;
  const int lane = tid & 31;
  const int w    = tid >> 5;                       // 0..15
  const int j    = blockIdx.x * 16 + w;            // 0..2047

  __bf16* wmy = wlds + (size_t)w * 4 * HD;

#if HAVE_TDM
  (void)Whh;
  {
    // D# group0: count=1, lds_addr, 57-bit global_addr, type=2
    const unsigned long long ga = (unsigned long long)(uintptr_t)(Whh_bf + (size_t)j * HD);
    const unsigned lds_off = (unsigned)(uintptr_t)(void*)wmy;   // low 32 bits = LDS byte addr
    uint4v g0;
    g0[0] = 1u;                                        // count=1, user mode
    g0[1] = lds_off;
    g0[2] = (unsigned)(ga & 0xFFFFFFFFu);
    g0[3] = (unsigned)((ga >> 32) & 0x1FFFFFFu) | (2u << 30);
    // D# group1: data_size=2B, tensor_dim0=2048, tensor_dim1=4, tile=2048x4,
    //            dim0 stride = HD*HD elements (gap between gate blocks)
    int8v g1;
    g1[0] = (int)(1u << 16);            // data_size code 1 (2 bytes)
    g1[1] = (int)(2048u << 16);         // tensor_dim0[15:0] @ bits 63:48
    g1[2] = (int)(4u << 16);            // tensor_dim1[15:0] @ bits 95:80
    g1[3] = (int)(2048u << 16);         // tile_dim0 @ bits 127:112
    g1[4] = 4;                          // tile_dim1 @ bits 143:128
    g1[5] = (int)(HD * HD);             // tensor_dim0_stride[31:0] = 4194304
    g1[6] = 0;
    g1[7] = 0;
    int4v gz;  gz[0] = 0; gz[1] = 0; gz[2] = 0; gz[3] = 0;
    int8v gz8; gz8[0] = 0; gz8[1] = 0; gz8[2] = 0; gz8[3] = 0;
    gz8[4] = 0; gz8[5] = 0; gz8[6] = 0; gz8[7] = 0;
    __builtin_amdgcn_tensor_load_to_lds(g0, g1, gz, gz, gz8, 0);   // 6-arg toolchain form
    __builtin_amdgcn_s_wait_tensorcnt(0);
  }
#else
  (void)Whh_bf;
#pragma unroll
  for (int r = 0; r < 4; ++r) {
    const float* src = Whh + (size_t)(r * HD + j) * HD;
    __bf16* dst = wmy + r * HD;
    for (int i = lane; i < HD / 4; i += 32) {
      float4 v = *(const float4*)(src + i * 4);
      bf16x4 p;
      p.x = to_bf16(v.x); p.y = to_bf16(v.y); p.z = to_bf16(v.z); p.w = to_bf16(v.w);
      *(bf16x4*)(dst + i * 4) = p;
    }
  }
#endif

  float c = c0[j];
  float h = h0[j];
  const unsigned nwg = gridDim.x;

  for (int t = 0; t < SL; ++t) {
    // Stage current h vector into LDS (8KB), cooperatively.
    const float* hsrc = (t == 0) ? h0 : (hbuf + (t & 1) * HD);
    for (int i = tid; i < HD / 4; i += LSTM_THR)
      *(float4*)(hsh + i * 4) = *(const float4*)(hsrc + i * 4);
    __syncthreads();

    // 4 dot products of length 2048, K split across 32 lanes.
    float a0 = 0.f, a1 = 0.f, a2 = 0.f, a3 = 0.f;
#pragma unroll 4
    for (int i = 0; i < 16; ++i) {
      const int k4 = lane + 32 * i;
      float4 hv = *(const float4*)(hsh + k4 * 4);
      bf16x4 w0 = *(const bf16x4*)(wmy + 0 * HD + k4 * 4);
      bf16x4 w1 = *(const bf16x4*)(wmy + 1 * HD + k4 * 4);
      bf16x4 w2 = *(const bf16x4*)(wmy + 2 * HD + k4 * 4);
      bf16x4 w3 = *(const bf16x4*)(wmy + 3 * HD + k4 * 4);
      a0 += (float)w0.x * hv.x + (float)w0.y * hv.y + (float)w0.z * hv.z + (float)w0.w * hv.w;
      a1 += (float)w1.x * hv.x + (float)w1.y * hv.y + (float)w1.z * hv.z + (float)w1.w * hv.w;
      a2 += (float)w2.x * hv.x + (float)w2.y * hv.y + (float)w2.z * hv.z + (float)w2.w * hv.w;
      a3 += (float)w3.x * hv.x + (float)w3.y * hv.y + (float)w3.z * hv.z + (float)w3.w * hv.w;
    }
#pragma unroll
    for (int off = 16; off; off >>= 1) {
      a0 += __shfl_xor(a0, off, 32);
      a1 += __shfl_xor(a1, off, 32);
      a2 += __shfl_xor(a2, off, 32);
      a3 += __shfl_xor(a3, off, 32);
    }

    const float* xg = xgates + (size_t)t * GD;
    const float gi = sigmoid_f(a0 + xg[0 * HD + j]);
    const float gf = sigmoid_f(a1 + xg[1 * HD + j]);
    const float gg = tanhf    (a2 + xg[2 * HD + j]);
    const float go = sigmoid_f(a3 + xg[3 * HD + j]);
    c = gf * c + gi * gg;
    h = go * tanhf(c);

    if (lane == 0) {
      hbuf[((t + 1) & 1) * HD + j] = h;            // next step's h (double buffer)
      hs_bf[(size_t)t * HD + j] = to_bf16(h);      // for output GEMM
    }
    __syncthreads();                                // protect hsh before next copy

    // Grid-wide barrier: release my h write, acquire everyone else's.
    __threadfence();
    if (tid == 0) {
      __hip_atomic_fetch_add(bar, 1u, __ATOMIC_ACQ_REL, __HIP_MEMORY_SCOPE_AGENT);
      const unsigned target = (unsigned)(t + 1) * nwg;
      while (__hip_atomic_load(bar, __ATOMIC_ACQUIRE, __HIP_MEMORY_SCOPE_AGENT) < target)
        __builtin_amdgcn_s_sleep(2);
    }
    __syncthreads();
  }

  if (lane == 0) {                                  // hL, cL tail of d_out
    out_tail[j]      = h;
    out_tail[HD + j] = c;
  }
}

// ---------------- fused relu + log_softmax (row of 256) ----------------
__global__ void k_logsoftmax(const float* __restrict__ logits, float* __restrict__ out) {
  const int rowi = blockIdx.x, tid = threadIdx.x;
  const int w = tid >> 5, lane = tid & 31;
  float r = fmaxf(logits[(size_t)rowi * NL + tid], 0.0f);

  float m = r;
#pragma unroll
  for (int off = 16; off; off >>= 1) m = fmaxf(m, __shfl_xor(m, off, 32));
  __shared__ float red[8];
  if (lane == 0) red[w] = m;
  __syncthreads();
  float mall = red[0];
#pragma unroll
  for (int i = 1; i < 8; ++i) mall = fmaxf(mall, red[i]);
  __syncthreads();

  float s = __expf(r - mall);
#pragma unroll
  for (int off = 16; off; off >>= 1) s += __shfl_xor(s, off, 32);
  if (lane == 0) red[w] = s;
  __syncthreads();
  float sall = 0.0f;
#pragma unroll
  for (int i = 0; i < 8; ++i) sall += red[i];

  out[(size_t)rowi * NL + tid] = r - mall - __logf(sall);
}

// ---------------- host launch ----------------
extern "C" void kernel_launch(void* const* d_in, const int* in_sizes, int n_in,
                              void* d_out, int out_size, void* d_ws, size_t ws_size,
                              hipStream_t stream) {
  (void)in_sizes; (void)n_in; (void)out_size; (void)ws_size;
  const int*   ids  = (const int*)d_in[0];
  const float* h0   = (const float*)d_in[1];
  const float* c0   = (const float*)d_in[2];
  const float* Wemb = (const float*)d_in[3];
  const float* Wih  = (const float*)d_in[4];
  const float* Whh  = (const float*)d_in[5];
  const float* bih  = (const float*)d_in[6];
  const float* bhh  = (const float*)d_in[7];
  const float* Wout = (const float*)d_in[8];
  const float* bout = (const float*)d_in[9];
  float* out = (float*)d_out;

  // workspace carve (256B aligned)
  char* ws = (char*)d_ws;
  size_t off = 0;
  auto take = [&](size_t bytes) -> char* {
    char* p = ws + off;
    off += (bytes + 255) & ~(size_t)255;
    return p;
  };
  unsigned* bar      = (unsigned*)take(256);
  float*    bias_sum = (float*)   take((size_t)GD * 4);
  __bf16*   Wih_bf   = (__bf16*)  take((size_t)GD * ED * 2);
  __bf16*   Whh_bf   = (__bf16*)  take((size_t)GD * HD * 2);
  __bf16*   Wout_bf  = (__bf16*)  take((size_t)NL * HD * 2);
  __bf16*   emb_bf   = (__bf16*)  take((size_t)SL * ED * 2);
  float*    xgates   = (float*)   take((size_t)SL * GD * 4);
  __bf16*   hs_bf    = (__bf16*)  take((size_t)SL * HD * 2);
  float*    hbuf     = (float*)   take((size_t)2 * HD * 4);
  float*    logits   = (float*)   take((size_t)SL * NL * 4);

  (void)hipMemsetAsync(bar, 0, sizeof(unsigned), stream);   // graph-capture-legal reset

  // prep: bias sum, bf16 weight copies, embedding gather
  k_bias_sum<<<GD / 256, 256, 0, stream>>>(bih, bhh, bias_sum);
  k_cvt_bf16<<<2048, 256, 0, stream>>>(Wih,  Wih_bf,  GD * ED);
  k_cvt_bf16<<<4096, 256, 0, stream>>>(Whh,  Whh_bf,  GD * HD);
  k_cvt_bf16<<<512,  256, 0, stream>>>(Wout, Wout_bf, NL * HD);
  k_embed<<<SL, 256, 0, stream>>>(ids, Wemb, emb_bf);

  // x_gates[SL,GD] = emb_bf16 @ W_ih^T + (b_ih + b_hh)
  dim3 g1(GD / BN, SL / BM);
  k_gemm_bf16<<<g1, 256, 0, stream>>>(xgates, emb_bf, Wih_bf, bias_sum, ED, ED, ED, GD);

  // sequential LSTM with LDS-resident W_hh (TDM-filled)
  (void)hipFuncSetAttribute((const void*)k_lstm,
                            hipFuncAttributeMaxDynamicSharedMemorySize, LSTM_LDS);
  k_lstm<<<LSTM_WGS, LSTM_THR, LSTM_LDS, stream>>>(Whh, Whh_bf, xgates, h0, c0, hbuf,
                                                   hs_bf, bar, out + (size_t)SL * NL);

  // logits[SL,NL] = hs_bf16 @ W_out^T + b_out
  dim3 g2(NL / BN, SL / BM);
  k_gemm_bf16<<<g2, 256, 0, stream>>>(logits, hs_bf, Wout_bf, bout, HD, HD, HD, NL);

  // out[SL,NL] = log_softmax(relu(logits))
  k_logsoftmax<<<SL, 256, 0, stream>>>(logits, out);
}
